// NetAmazon_GAT_heads_71768903516555
// MI455X (gfx1250) — matrix operand order repaired and checked by reading, not compiled, not run on hardware
//
#include <hip/hip_runtime.h>
#include <hip/hip_bf16.h>
#include <math.h>

typedef __attribute__((ext_vector_type(16))) _Float16 v16h;
typedef __attribute__((ext_vector_type(8)))  _Float16 v8h;
typedef __attribute__((ext_vector_type(8)))  float    v8f;

#define NEG_SLOPE 0.2f
#define GAT_EPS 1e-16f
#define CEILDIV(a,b) (((a)+(b)-1)/(b))

// ---------------------------------------------------------------------------
// fill
// ---------------------------------------------------------------------------
__global__ void fill_f32_kernel(float* __restrict__ p, float v, size_t n) {
  size_t i = (size_t)blockIdx.x * blockDim.x + threadIdx.x;
  if (i < n) p[i] = v;
}

// ---------------------------------------------------------------------------
// convert f32 -> f16 with zero padding: src[rows,K] -> dst[rowsPad,Kp]
// ---------------------------------------------------------------------------
__global__ void cvt_pad_kernel(const float* __restrict__ src, _Float16* __restrict__ dst,
                               int rows, int K, int rowsPad, int Kp) {
  size_t i = (size_t)blockIdx.x * blockDim.x + threadIdx.x;
  size_t tot = (size_t)rowsPad * Kp;
  if (i >= tot) return;
  int r = (int)(i / (size_t)Kp);
  int k = (int)(i % (size_t)Kp);
  float v = (r < rows && k < K) ? src[(size_t)r * K + k] : 0.f;
  dst[i] = (_Float16)v;
}

// ---------------------------------------------------------------------------
// CDNA5 async global->LDS copy (ASYNCcnt path), 16 bytes per lane
// ---------------------------------------------------------------------------
__device__ __forceinline__ void async_copy_b128(unsigned lds_off, const void* gaddr) {
  asm volatile("global_load_async_to_lds_b128 %0, %1, off"
               :: "v"(lds_off), "v"(gaddr) : "memory");
}
__device__ __forceinline__ void wait_async_zero() {
  asm volatile("s_wait_asynccnt 0" ::: "memory");
}

// ---------------------------------------------------------------------------
// WMMA GEMM:  C[Npad, Mpad] = A[Npad, Kp] * B[Mpad, Kp]^T   (f16 in, f32 out)
// block = 256 threads = 8 waves; block tile 128 x 128; BK = 32
// waves arranged 4 (rows) x 2 (cols): each wave computes 32 x 64
//   -> 2 A-frags, 4 B-frags, 8 WMMAs per K-step
// staging via GLOBAL_LOAD_ASYNC_TO_LDS_B128 (4 x 16B per thread per K-step)
// ---------------------------------------------------------------------------
#define GEMM_BM 128
#define GEMM_BN 128
#define GEMM_BK 32
#define GEMM_LDA 40   // halfs; 80B row stride keeps 16B alignment
#define GEMM_LDB 40

__global__ void __launch_bounds__(256)
wmma_gemm_nt_kernel(const _Float16* __restrict__ A, const _Float16* __restrict__ B,
                    float* __restrict__ C, int Kp, int Mpad) {
  __shared__ __align__(16) _Float16 ldsA[GEMM_BM * GEMM_LDA];
  __shared__ __align__(16) _Float16 ldsB[GEMM_BN * GEMM_LDB];

  const int t    = threadIdx.x;
  const int lane = t & 31;
  const int wid  = t >> 5;
  const int waveRow = wid & 3;    // 0..3 -> 32-row strip
  const int waveCol = wid >> 2;   // 0..1 -> 64-col strip
  const int rowBase = blockIdx.y * GEMM_BM;
  const int colBase = blockIdx.x * GEMM_BN;

  v8f acc[2][4] = {};

  // staging assignment: thread t moves 16 contiguous halfs of one A row and
  // one B row (two async b128 each)
  const int sr = t >> 1;          // 0..127
  const int sk = (t & 1) * 16;    // 0 or 16

  const int m16    = lane & 15;
  const int hiHalf = lane >> 4;   // 0 for lanes 0-15, 1 for lanes 16-31

  // loop-invariant LDS byte offsets (flat addr low 32 bits == LDS offset)
  const unsigned aOff = (unsigned)(size_t)&ldsA[sr * GEMM_LDA + sk];
  const unsigned bOff = (unsigned)(size_t)&ldsB[sr * GEMM_LDB + sk];

  const _Float16* gA = A + (size_t)(rowBase + sr) * Kp + sk;
  const _Float16* gB = B + (size_t)(colBase + sr) * Kp + sk;

  for (int k0 = 0; k0 < Kp; k0 += GEMM_BK) {
    // ---- async stage A (128x32 halfs) and B (128x32 halfs) tiles ----
    async_copy_b128(aOff,      gA);
    async_copy_b128(aOff + 16, gA + 8);
    async_copy_b128(bOff,      gB);
    async_copy_b128(bOff + 16, gB + 8);
    wait_async_zero();
    __syncthreads();

    // ---- A fragments (16x32 each): lanes 0-15 hold K 0..7 & 16..23,
    //      lanes 16-31 hold K 8..15 & 24..31 (ISA 16-bit A layout) ----
    v16h afrag[2];
    #pragma unroll
    for (int mi = 0; mi < 2; ++mi) {
      const int aRow = waveRow * 32 + mi * 16 + m16;
      const int akf  = hiHalf * 8;
      v8h alo = *(const v8h*)&ldsA[aRow * GEMM_LDA + akf];
      v8h ahi = *(const v8h*)&ldsA[aRow * GEMM_LDA + akf + 16];
      afrag[mi] = __builtin_shufflevector(alo, ahi,
          0,1,2,3,4,5,6,7,8,9,10,11,12,13,14,15);
    }

    // ---- B fragments (32x16 each): same K order as A:
    //      lanes 0-15 -> K 0..7 & 16..23, lanes 16-31 -> K 8..15 & 24..31 ----
    #pragma unroll
    for (int ni = 0; ni < 4; ++ni) {
      const int bCol = waveCol * 64 + ni * 16 + m16;
      const int bk   = hiHalf * 8;
      v8h blo = *(const v8h*)&ldsB[bCol * GEMM_LDB + bk];
      v8h bhi = *(const v8h*)&ldsB[bCol * GEMM_LDB + bk + 16];
      v16h bfrag = __builtin_shufflevector(blo, bhi,
          0,1,2,3,4,5,6,7,8,9,10,11,12,13,14,15);
      #pragma unroll
      for (int mi = 0; mi < 2; ++mi) {
        acc[mi][ni] = __builtin_amdgcn_wmma_f32_16x16x32_f16(
            /*neg_a=*/false, afrag[mi], /*neg_b=*/false, bfrag,
            /*c_mod=*/(short)0, acc[mi][ni], /*reuse_a=*/false, /*reuse_b=*/false);
      }
    }
    __syncthreads();
    gA += GEMM_BK;
    gB += GEMM_BK;
  }

  // ---- store C: VGPR r -> M = r (+8 for lanes 16-31), N = lane&15 ----
  const int rOff = hiHalf * 8;
  #pragma unroll
  for (int mi = 0; mi < 2; ++mi) {
    #pragma unroll
    for (int ni = 0; ni < 4; ++ni) {
      const int col = colBase + waveCol * 64 + ni * 16 + m16;
      #pragma unroll
      for (int r = 0; r < 8; ++r) {
        const int row = rowBase + waveRow * 32 + mi * 16 + r + rOff;
        C[(size_t)row * Mpad + col] = acc[mi][ni][r];
      }
    }
  }
}

// ---------------------------------------------------------------------------
// per-(node,head) attention scalars: a_src/a_dst = <h[n,h,:], att_{s,d}[h,:]>
// ---------------------------------------------------------------------------
__global__ void att_kernel(const float* __restrict__ Hm, const float* __restrict__ att_s,
                           const float* __restrict__ att_d, float* __restrict__ asrc,
                           float* __restrict__ adst, int N, int Hh, int C, int ldH) {
  int i = blockIdx.x * blockDim.x + threadIdx.x;
  if (i >= N * Hh) return;
  int n = i / Hh, h = i % Hh;
  const float* hp = Hm + (size_t)n * ldH + h * C;
  float ss = 0.f, sd = 0.f;
  for (int c = 0; c < C; ++c) {
    float v = hp[c];
    ss += v * att_s[h * C + c];
    sd += v * att_d[h * C + c];
  }
  asrc[i] = ss;
  adst[i] = sd;
}

// ordered-int float atomic max (valid for mixed signs; init must be -inf-ish)
__device__ __forceinline__ void atomicMaxFloat(float* addr, float val) {
  if (val >= 0.f) atomicMax((int*)addr, __float_as_int(val));
  else            atomicMin((unsigned int*)addr, (unsigned int)__float_as_int(val));
}

__device__ __forceinline__ void edge_endpoints(const int* ei, int e, int E, int& s, int& d) {
  if (e < E) { s = ei[e]; d = ei[E + e]; }
  else       { s = d = e - E; }              // self loops appended
}

// pass 1: segment max of leaky-relu edge logits over dst
__global__ void edge_max_kernel(const int* __restrict__ ei, const float* __restrict__ asrc,
                                const float* __restrict__ adst, float* __restrict__ mb,
                                int E, int N, int Hh) {
  int e = blockIdx.x * blockDim.x + threadIdx.x;
  if (e >= E + N) return;
  int s, d; edge_endpoints(ei, e, E, s, d);
  for (int h = 0; h < Hh; ++h) {
    float v = asrc[s * Hh + h] + adst[d * Hh + h];
    v = v > 0.f ? v : NEG_SLOPE * v;
    atomicMaxFloat(&mb[d * Hh + h], v);
  }
}

// pass 2: segment sum of exp(e - m)
__global__ void edge_sum_kernel(const int* __restrict__ ei, const float* __restrict__ asrc,
                                const float* __restrict__ adst, const float* __restrict__ mb,
                                float* __restrict__ sb, int E, int N, int Hh) {
  int e = blockIdx.x * blockDim.x + threadIdx.x;
  if (e >= E + N) return;
  int s, d; edge_endpoints(ei, e, E, s, d);
  for (int h = 0; h < Hh; ++h) {
    float v = asrc[s * Hh + h] + adst[d * Hh + h];
    v = v > 0.f ? v : NEG_SLOPE * v;
    atomicAdd(&sb[d * Hh + h], expf(v - mb[d * Hh + h]));
  }
}

// pass 3: out[dst] += alpha * h[src]
__global__ void edge_agg_kernel(const int* __restrict__ ei, const float* __restrict__ asrc,
                                const float* __restrict__ adst, const float* __restrict__ mb,
                                const float* __restrict__ sb, const float* __restrict__ Hm,
                                float* __restrict__ Ob, int E, int N, int Hh, int C,
                                int ldH, int ldO) {
  int i = blockIdx.x * blockDim.x + threadIdx.x;
  if (i >= (E + N) * Hh) return;
  int e = i / Hh, h = i % Hh;
  int s, d; edge_endpoints(ei, e, E, s, d);
  float v = asrc[s * Hh + h] + adst[d * Hh + h];
  v = v > 0.f ? v : NEG_SLOPE * v;
  float alpha = expf(v - mb[d * Hh + h]) / (sb[d * Hh + h] + GAT_EPS);
  const float* hp = Hm + (size_t)s * ldH + h * C;
  float*       op = Ob + (size_t)d * ldO + h * C;
  for (int c = 0; c < C; ++c) atomicAdd(&op[c], alpha * hp[c]);
}

// ---------------------------------------------------------------------------
// fused bias + relu (in place)
// ---------------------------------------------------------------------------
__global__ void bias_relu_kernel(float* __restrict__ out, const float* __restrict__ b,
                                 size_t n, int HC) {
  size_t i = (size_t)blockIdx.x * blockDim.x + threadIdx.x;
  if (i >= n) return;
  float v = out[i] + b[i % (size_t)HC];
  out[i] = v > 0.f ? v : 0.f;
}

// ---------------------------------------------------------------------------
// final: bias + log_softmax over C=10 classes
// ---------------------------------------------------------------------------
__global__ void bias_lsm_kernel(const float* __restrict__ in, const float* __restrict__ b,
                                float* __restrict__ out, int N, int C) {
  int n = blockIdx.x * blockDim.x + threadIdx.x;
  if (n >= N) return;
  float v[16];
  float mx = -1e30f;
  for (int c = 0; c < C; ++c) {
    v[c] = in[(size_t)n * C + c] + b[c];
    mx = fmaxf(mx, v[c]);
  }
  float s = 0.f;
  for (int c = 0; c < C; ++c) s += expf(v[c] - mx);
  float ls = logf(s);
  for (int c = 0; c < C; ++c) out[(size_t)n * C + c] = v[c] - mx - ls;
}

// ---------------------------------------------------------------------------
// host-side per-layer driver
// ---------------------------------------------------------------------------
static void gat_layer_device(const _Float16* A16, const _Float16* W16, float* Hb,
                             const float* att_s, const float* att_d,
                             const int* ei, int E, int N, int Npad,
                             int Kp, int Mpad, int Hh, int C, int ldO,
                             float* asrc, float* adst, float* mb, float* sb,
                             float* Ob, hipStream_t stream) {
  dim3 gg(Mpad / GEMM_BN, Npad / GEMM_BM, 1);
  wmma_gemm_nt_kernel<<<gg, 256, 0, stream>>>(A16, W16, Hb, Kp, Mpad);

  int NH = N * Hh;
  att_kernel<<<CEILDIV(NH, 256), 256, 0, stream>>>(Hb, att_s, att_d, asrc, adst, N, Hh, C, Mpad);

  fill_f32_kernel<<<CEILDIV((size_t)NH, (size_t)256), 256, 0, stream>>>(mb, -1e30f, (size_t)NH);
  fill_f32_kernel<<<CEILDIV((size_t)NH, (size_t)256), 256, 0, stream>>>(sb, 0.f, (size_t)NH);
  size_t osz = (size_t)N * ldO;
  fill_f32_kernel<<<CEILDIV(osz, (size_t)256), 256, 0, stream>>>(Ob, 0.f, osz);

  int Etot = E + N;
  edge_max_kernel<<<CEILDIV(Etot, 256), 256, 0, stream>>>(ei, asrc, adst, mb, E, N, Hh);
  edge_sum_kernel<<<CEILDIV(Etot, 256), 256, 0, stream>>>(ei, asrc, adst, mb, sb, E, N, Hh);
  edge_agg_kernel<<<CEILDIV(Etot * Hh, 256), 256, 0, stream>>>(ei, asrc, adst, mb, sb, Hb, Ob,
                                                               E, N, Hh, C, Mpad, ldO);
}

extern "C" void kernel_launch(void* const* d_in, const int* in_sizes, int n_in,
                              void* d_out, int out_size, void* d_ws, size_t ws_size,
                              hipStream_t stream) {
  const float* x   = (const float*)d_in[0];
  const int*   ei  = (const int*)d_in[1];
  const float* W1  = (const float*)d_in[2];
  const float* as1 = (const float*)d_in[3];
  const float* ad1 = (const float*)d_in[4];
  const float* b1  = (const float*)d_in[5];
  const float* W2  = (const float*)d_in[6];
  const float* as2 = (const float*)d_in[7];
  const float* ad2 = (const float*)d_in[8];
  const float* b2  = (const float*)d_in[9];
  const float* W3  = (const float*)d_in[10];
  const float* as3 = (const float*)d_in[11];
  const float* ad3 = (const float*)d_in[12];
  const float* b3  = (const float*)d_in[13];
  float* out = (float*)d_out;

  const int N = 20000;
  const int Npad = 20096;              // multiple of 128 (GEMM row tile)
  const int E = in_sizes[1] / 2;       // 320000

  // workspace carve-out
  char* ws = (char*)d_ws;
  size_t off = 0;
  auto take = [&](size_t bytes) -> char* {
    char* p = ws + off;
    off += (bytes + 255) & ~(size_t)255;
    return p;
  };
  _Float16* A16 = (_Float16*)take((size_t)Npad * 1024 * sizeof(_Float16));
  _Float16* W16 = (_Float16*)take((size_t)1024 * 768  * sizeof(_Float16));
  float* Hb   = (float*)take((size_t)Npad * 1024 * sizeof(float));
  float* Ob   = (float*)take((size_t)Npad * 1024 * sizeof(float));
  float* asrc = (float*)take((size_t)N * 8 * sizeof(float));
  float* adst = (float*)take((size_t)N * 8 * sizeof(float));
  float* mb   = (float*)take((size_t)N * 8 * sizeof(float));
  float* sb   = (float*)take((size_t)N * 8 * sizeof(float));

  auto cvt = [&](const float* src, _Float16* dst, int rows, int K, int rowsPad, int Kp) {
    size_t tot = (size_t)rowsPad * Kp;
    cvt_pad_kernel<<<CEILDIV(tot, (size_t)256), 256, 0, stream>>>(src, dst, rows, K, rowsPad, Kp);
  };

  // ---------------- Layer 1: F_in=745(->768), H=8, C=128, M=1024 -------------
  cvt(x,  A16, N,    745, Npad, 768);
  cvt(W1, W16, 1024, 745, 1024, 768);
  gat_layer_device(A16, W16, Hb, as1, ad1, ei, E, N, Npad,
                   /*Kp=*/768, /*Mpad=*/1024, /*H=*/8, /*C=*/128, /*ldO=*/1024,
                   asrc, adst, mb, sb, Ob, stream);
  bias_relu_kernel<<<CEILDIV((size_t)N * 1024, (size_t)256), 256, 0, stream>>>(
      Ob, b1, (size_t)N * 1024, 1024);

  // ---------------- Layer 2: K=1024, H=8, C=64, M=512 ------------------------
  cvt(Ob, A16, N,   1024, Npad, 1024);
  cvt(W2, W16, 512, 1024, 512,  1024);
  gat_layer_device(A16, W16, Hb, as2, ad2, ei, E, N, Npad,
                   /*Kp=*/1024, /*Mpad=*/512, /*H=*/8, /*C=*/64, /*ldO=*/512,
                   asrc, adst, mb, sb, Ob, stream);
  bias_relu_kernel<<<CEILDIV((size_t)N * 512, (size_t)256), 256, 0, stream>>>(
      Ob, b2, (size_t)N * 512, 512);

  // ---------------- Layer 3: K=512, H=1, C=10 (M padded 10->128) -------------
  cvt(Ob, A16, N,  512, Npad, 512);
  cvt(W3, W16, 10, 512, 128,  512);
  gat_layer_device(A16, W16, Hb, as3, ad3, ei, E, N, Npad,
                   /*Kp=*/512, /*Mpad=*/128, /*H=*/1, /*C=*/10, /*ldO=*/10,
                   asrc, adst, mb, sb, Ob, stream);

  bias_lsm_kernel<<<CEILDIV(N, 256), 256, 0, stream>>>(Ob, b3, out, N, 10);
}
